// SpectralLanguageGenerator_48954037240519
// MI455X (gfx1250) — matrix operand order repaired and checked by reading, not compiled, not run on hardware
//
#include <hip/hip_runtime.h>
#include <hip/hip_bf16.h>
#include <math.h>

// ---------------------------------------------------------------------------
// Sizes fixed by reference: B=2,S=1024 -> M=2048 rows; D=512; C=1000(->1024
// padded); V=50000; MLP 1024->512->512->768->50000.
// ---------------------------------------------------------------------------
#define MROWS   2048
#define DDIM    512
#define CCON    1000
#define CPAD    1024
#define VOCAB   50000
#define TOPK    5
#define EPS_CS  1e-8f

typedef __attribute__((ext_vector_type(16))) __bf16 v16bf;
typedef __attribute__((ext_vector_type(8)))  float  v8f;
typedef int async_i4 __attribute__((vector_size(16)));   // b128 payload type

// CDNA5 async global->LDS copies (ASYNCcnt-tracked), guarded so both host
// pass and older toolchains still compile via the synchronous fallback.
// Probe-verified signature: (AS1 v4i*, AS3 v4i*, imm offset, imm cpol).
#if defined(__HIP_DEVICE_COMPILE__) &&                                        \
    __has_builtin(__builtin_amdgcn_global_load_async_to_lds_b128) &&          \
    __has_builtin(__builtin_amdgcn_s_wait_asynccnt)
#define USE_ASYNC_LDS 1
#define ASYNC_CP_B128(gsrc, ldst)                                             \
    __builtin_amdgcn_global_load_async_to_lds_b128(                           \
        (__attribute__((address_space(1))) async_i4*)(gsrc),                  \
        (__attribute__((address_space(3))) async_i4*)(ldst), 0, 0)
#define WAIT_ASYNC() __builtin_amdgcn_s_wait_asynccnt(0)
#else
#define USE_ASYNC_LDS 0
#define ASYNC_CP_B128(gsrc, ldst)                                             \
    (*reinterpret_cast<uint4*>(ldst) =                                        \
         *reinterpret_cast<const uint4*>(gsrc))
#define WAIT_ASYNC() ((void)0)
#endif

// ---------------------------------------------------------------------------
// Kernel 1: magnitudes + concat(real,imag) in bf16, row norms xn.
// ---------------------------------------------------------------------------
__global__ __launch_bounds__(256)
void prep_spectrum_kernel(const float* __restrict__ sr, const float* __restrict__ si,
                          __bf16* __restrict__ magb, __bf16* __restrict__ xcat,
                          float* __restrict__ xn) {
    __shared__ float red[256];
    const int m = blockIdx.x;
    const int t = threadIdx.x;
    float acc = 0.f;
#pragma unroll
    for (int j = 0; j < 2; ++j) {
        int c = t + 256 * j;
        float a = sr[(size_t)m * DDIM + c];
        float b = si[(size_t)m * DDIM + c];
        float mg = sqrtf(a * a + b * b);
        magb[(size_t)m * DDIM + c] = (__bf16)mg;
        xcat[(size_t)m * (2 * DDIM) + c]        = (__bf16)a;
        xcat[(size_t)m * (2 * DDIM) + DDIM + c] = (__bf16)b;
        acc += a * a + b * b;   // == mg*mg
    }
    red[t] = acc;
    __syncthreads();
    for (int s = 128; s > 0; s >>= 1) {
        if (t < s) red[t] += red[t + s];
        __syncthreads();
    }
    if (t == 0) xn[m] = sqrtf(red[0]);
}

// ---------------------------------------------------------------------------
// Kernel 2: concept magnitudes (bf16) + norms yn, rows >= 1000 zero-padded.
// ---------------------------------------------------------------------------
__global__ __launch_bounds__(256)
void prep_concepts_kernel(const float* __restrict__ cr, const float* __restrict__ ci,
                          __bf16* __restrict__ cmagb, float* __restrict__ yn) {
    __shared__ float red[256];
    const int c = blockIdx.x;       // 0..1023
    const int t = threadIdx.x;
    float acc = 0.f;
#pragma unroll
    for (int j = 0; j < 2; ++j) {
        int d = t + 256 * j;
        float mg = 0.f;
        if (c < CCON) {
            float a = cr[(size_t)c * DDIM + d];
            float b = ci[(size_t)c * DDIM + d];
            mg = sqrtf(a * a + b * b);
            acc += a * a + b * b;
        }
        cmagb[(size_t)c * DDIM + d] = (__bf16)mg;
    }
    red[t] = acc;
    __syncthreads();
    for (int s = 128; s > 0; s >>= 1) {
        if (t < s) red[t] += red[t + s];
        __syncthreads();
    }
    if (t == 0) yn[c] = (c < CCON) ? sqrtf(red[0]) : 0.f;
}

// ---------------------------------------------------------------------------
// bf16 WMMA GEMM, 128(M) x 64(N) block tile, BK=64, double-buffered LDS with
// async global->LDS copies. 8 waves in a 4(M)x2(N) grid; each wave owns a
// 2x2 grid of 16x16 f32 accumulators -> 8 v_wmma per stage vs 16 ds_load_b128.
//   BMODE 0: B is f32 [K,N] row-major (weights), float4 loads + bf16 cvt.
//   BMODE 1: B is bf16 [N,K] row-major (B^T layout; cmag case), async-copied.
//   EPI 0: v / (max(xn[m],eps)*max(yn[n],eps)) -> f32 nontemporal
//   EPI 1: gelu(v + bias[n]) -> bf16
//   EPI 2: v + bias[n] -> f32 nontemporal (logits), n < N guarded
// ---------------------------------------------------------------------------
template<int BMODE, int EPI>
__global__ __launch_bounds__(256)
void gemm_bf16_wmma_kernel(const __bf16* __restrict__ A,
                           const float*  __restrict__ Bf,
                           const __bf16* __restrict__ Bb,
                           float*  __restrict__ Cf,
                           __bf16* __restrict__ Cb,
                           const float* __restrict__ bias,
                           const float* __restrict__ rown,
                           const float* __restrict__ coln,
                           int M, int N, int K, int ldb, int ldc) {
    constexpr int LDA = 72;   // bf16 elems per A-LDS row (64 + 8 pad, 144B)
    constexpr int LDB = 72;   // bf16 elems per B-LDS row ([n][k] layout)
    __shared__ __bf16 As[2][128 * LDA];
    __shared__ __bf16 Bs[2][64 * LDB];

    const int m0 = blockIdx.y * 128;
    const int n0 = blockIdx.x * 64;
    const int t    = threadIdx.x;
    const int lane = t & 31;
    const int w    = t >> 5;
    const int mwb  = (w & 3) * 32;   // wave's M base inside tile
    const int nwb  = (w >> 2) * 32;  // wave's N base inside tile

    v8f acc[2][2];
#pragma unroll
    for (int mi = 0; mi < 2; ++mi)
#pragma unroll
        for (int ni = 0; ni < 2; ++ni)
            acc[mi][ni] = v8f{0.f, 0.f, 0.f, 0.f, 0.f, 0.f, 0.f, 0.f};

    const int kh = lane >> 4;   // K-half of the wave per ISA bf16 layouts
    const int l15 = lane & 15;

    // ---- stage loader: global -> LDS buffer `buf` for K-slice [k0,k0+64) --
    auto load_stage = [&](int buf, int k0) {
        // A tile 128x64 bf16: 1024 16B chunks, 4 per thread, async.
#pragma unroll
        for (int j = 0; j < 4; ++j) {
            int q   = t + 256 * j;
            int row = q >> 3;
            int c8  = (q & 7) * 8;
            const __bf16* gsrc = A + (size_t)(m0 + row) * K + k0 + c8;
            ASYNC_CP_B128(gsrc, &As[buf][row * LDA + c8]);
        }
        if (BMODE == 1) {
            // B^T tile 64x64 bf16: 512 chunks, 2 per thread, async.
#pragma unroll
            for (int j = 0; j < 2; ++j) {
                int q   = t + 256 * j;
                int row = q >> 3;
                int c8  = (q & 7) * 8;
                const __bf16* gsrc = Bb + (size_t)(n0 + row) * ldb + k0 + c8;
                ASYNC_CP_B128(gsrc, &Bs[buf][row * LDB + c8]);
            }
        } else {
            // f32 weights [K,N]: float4 loads (coalesced over n), cvt bf16,
            // transpose-store into [n][k]. Thread t -> n-group (t&15)*4,
            // k = (t>>4) + 16*j, so each (n,k) covered exactly once.
            const int nb = (t & 15) * 4;
            const int kb = t >> 4;                 // 0..15
            const bool full = (n0 + nb + 3) < N;
            if (k0 + 64 < K)
                __builtin_prefetch(&Bf[(size_t)(k0 + 64 + kb) * N + n0 + nb], 0, 1);
#pragma unroll
            for (int j = 0; j < 4; ++j) {
                int k = kb + 16 * j;
                const float* src = &Bf[(size_t)(k0 + k) * N + n0 + nb];
                float4 f;
                if (full) {
                    f = *reinterpret_cast<const float4*>(src);
                } else {
                    f.x = (n0 + nb + 0 < N) ? src[0] : 0.f;
                    f.y = (n0 + nb + 1 < N) ? src[1] : 0.f;
                    f.z = (n0 + nb + 2 < N) ? src[2] : 0.f;
                    f.w = (n0 + nb + 3 < N) ? src[3] : 0.f;
                }
                Bs[buf][(nb + 0) * LDB + k] = (__bf16)f.x;
                Bs[buf][(nb + 1) * LDB + k] = (__bf16)f.y;
                Bs[buf][(nb + 2) * LDB + k] = (__bf16)f.z;
                Bs[buf][(nb + 3) * LDB + k] = (__bf16)f.w;
            }
        }
    };

    // ---- one stage of WMMAs on buffer `buf` ----
    auto compute_stage = [&](int buf) {
#pragma unroll
        for (int kk = 0; kk < 64; kk += 32) {
            union Frag { uint4 u[2]; v16bf v; };
            Frag fa[2], fb[2];
#pragma unroll
            for (int mi = 0; mi < 2; ++mi) {
                const __bf16* p = &As[buf][(mwb + mi * 16 + l15) * LDA + kk + 8 * kh];
                fa[mi].u[0] = *reinterpret_cast<const uint4*>(p);
                fa[mi].u[1] = *reinterpret_cast<const uint4*>(p + 16);
            }
#pragma unroll
            for (int ni = 0; ni < 2; ++ni) {
                const __bf16* p = &Bs[buf][(nwb + ni * 16 + l15) * LDB + kk + 8 * kh];
                fb[ni].u[0] = *reinterpret_cast<const uint4*>(p);
                fb[ni].u[1] = *reinterpret_cast<const uint4*>(p + 16);
            }
#pragma unroll
            for (int mi = 0; mi < 2; ++mi)
#pragma unroll
                for (int ni = 0; ni < 2; ++ni)
                    acc[mi][ni] = __builtin_amdgcn_wmma_f32_16x16x32_bf16(
                        false, fa[mi].v, false, fb[ni].v, (short)0,
                        acc[mi][ni], false, false);
        }
    };

    // ---- software pipeline: copy(s+1) overlaps compute(s) ----
    const int stages = K >> 6;
    load_stage(0, 0);
    WAIT_ASYNC();
    __syncthreads();
    for (int s = 0; s < stages; ++s) {
        const int cur = s & 1;
        if (s + 1 < stages) load_stage(cur ^ 1, (s + 1) * 64);
        compute_stage(cur);
        WAIT_ASYNC();
        __syncthreads();
    }

    // ---- epilogue: C/D layout — VGPR r: lanes0-15 M=r, lanes16-31 M=8+r ---
    const int mh = lane >> 4;
#pragma unroll
    for (int mi = 0; mi < 2; ++mi) {
#pragma unroll
        for (int ni = 0; ni < 2; ++ni) {
            int n = n0 + nwb + ni * 16 + l15;
#pragma unroll
            for (int r = 0; r < 8; ++r) {
                int m = m0 + mwb + mi * 16 + mh * 8 + r;
                float v = acc[mi][ni][r];
                if (EPI == 0) {
                    float d = v / (fmaxf(rown[m], EPS_CS) * fmaxf(coln[n], EPS_CS));
                    __builtin_nontemporal_store(d, &Cf[(size_t)m * ldc + n]);
                } else if (EPI == 1) {
                    float x = v + bias[n];
                    float g = 0.5f * x * (1.f + erff(x * 0.70710678118654752f));
                    Cb[(size_t)m * ldc + n] = (__bf16)g;
                } else {
                    if (n < N) {
                        float x = v + bias[n];
                        __builtin_nontemporal_store(x, &Cf[(size_t)m * ldc + n]);
                    }
                }
            }
        }
    }
}

// ---------------------------------------------------------------------------
// Per-row top-5 over 1000 sims (block per row, 5 masked argmaxes).
// ---------------------------------------------------------------------------
__global__ __launch_bounds__(256)
void topk_kernel(const float* __restrict__ sims,
                 float* __restrict__ out_sims, int* __restrict__ out_idx) {
    __shared__ float sv[256];
    __shared__ int   si_[256];
    const int row = blockIdx.x;
    const int t   = threadIdx.x;
    float vals[4];
    int   cols[4];
#pragma unroll
    for (int j = 0; j < 4; ++j) {
        int c = t + 256 * j;
        cols[j] = c;
        vals[j] = (c < CCON) ? sims[(size_t)row * CPAD + c] : -INFINITY;
    }
    for (int kk = 0; kk < TOPK; ++kk) {
        float bv = -INFINITY; int bi = 0x7fffffff;
#pragma unroll
        for (int j = 0; j < 4; ++j) {
            if (vals[j] > bv || (vals[j] == bv && cols[j] < bi)) {
                bv = vals[j]; bi = cols[j];
            }
        }
        sv[t] = bv; si_[t] = bi;
        __syncthreads();
        for (int s = 128; s > 0; s >>= 1) {
            if (t < s) {
                if (sv[t + s] > sv[t] ||
                    (sv[t + s] == sv[t] && si_[t + s] < si_[t])) {
                    sv[t] = sv[t + s]; si_[t] = si_[t + s];
                }
            }
            __syncthreads();
        }
        float wv = sv[0]; int wi = si_[0];
        if (t == 0) {
            out_sims[row * TOPK + kk] = wv;
            out_idx[row * TOPK + kk]  = wi;
        }
#pragma unroll
        for (int j = 0; j < 4; ++j)
            if (cols[j] == wi) vals[j] = -INFINITY;
        __syncthreads();
    }
}

// ---------------------------------------------------------------------------
// Host-side launcher.
// ---------------------------------------------------------------------------
extern "C" void kernel_launch(void* const* d_in, const int* in_sizes, int n_in,
                              void* d_out, int out_size, void* d_ws, size_t ws_size,
                              hipStream_t stream) {
    const float* sr = (const float*)d_in[0];
    const float* si = (const float*)d_in[1];
    const float* cr = (const float*)d_in[2];
    const float* ci = (const float*)d_in[3];
    const float* W1 = (const float*)d_in[4];
    const float* b1 = (const float*)d_in[5];
    const float* W2 = (const float*)d_in[6];
    const float* b2 = (const float*)d_in[7];
    const float* W3 = (const float*)d_in[8];
    const float* b3 = (const float*)d_in[9];
    const float* W4 = (const float*)d_in[10];
    const float* b4 = (const float*)d_in[11];

    // workspace carve-up (all chunks 4 KB aligned)
    char* p = (char*)d_ws;
    __bf16* magb  = (__bf16*)p; p += (size_t)MROWS * DDIM * 2;       // 2 MB
    __bf16* cmagb = (__bf16*)p; p += (size_t)CPAD  * DDIM * 2;       // 1 MB
    __bf16* xcat  = (__bf16*)p; p += (size_t)MROWS * 2 * DDIM * 2;   // 4 MB
    __bf16* h1    = (__bf16*)p; p += (size_t)MROWS * 512 * 2;        // 2 MB
    __bf16* h2    = (__bf16*)p; p += (size_t)MROWS * 512 * 2;        // 2 MB
    __bf16* h3    = (__bf16*)p; p += (size_t)MROWS * 768 * 2;        // 3 MB
    float*  xn    = (float*)p;  p += (size_t)MROWS * 4;              // 8 KB
    float*  yn    = (float*)p;  p += (size_t)CPAD * 4;               // 4 KB
    float*  simsb = (float*)p;  p += (size_t)MROWS * CPAD * 4;       // 8 MB

    // output layout: [top_sims f32 | top_idx i32 | logits f32]
    float* out_sims = (float*)d_out;
    int*   out_idx  = (int*)((float*)d_out + (size_t)MROWS * TOPK);
    float* logits   = (float*)d_out + (size_t)MROWS * TOPK * 2;

    prep_spectrum_kernel<<<MROWS, 256, 0, stream>>>(sr, si, magb, xcat, xn);
    prep_concepts_kernel<<<CPAD, 256, 0, stream>>>(cr, ci, cmagb, yn);

    // cosine-sim GEMM: [2048,512] x [512,1024-padded] -> sims f32
    gemm_bf16_wmma_kernel<1, 0><<<dim3(CPAD / 64, MROWS / 128), 256, 0, stream>>>(
        magb, nullptr, cmagb, simsb, nullptr, nullptr, xn, yn,
        MROWS, CPAD, DDIM, DDIM, CPAD);

    topk_kernel<<<MROWS, 256, 0, stream>>>(simsb, out_sims, out_idx);

    // MLP hidden layers with fused exact GELU, bf16 activations
    gemm_bf16_wmma_kernel<0, 1><<<dim3(512 / 64, MROWS / 128), 256, 0, stream>>>(
        xcat, W1, nullptr, nullptr, h1, b1, nullptr, nullptr,
        MROWS, 512, 2 * DDIM, 0, 512);
    gemm_bf16_wmma_kernel<0, 1><<<dim3(512 / 64, MROWS / 128), 256, 0, stream>>>(
        h1, W2, nullptr, nullptr, h2, b2, nullptr, nullptr,
        MROWS, 512, 512, 0, 512);
    gemm_bf16_wmma_kernel<0, 1><<<dim3(768 / 64, MROWS / 128), 256, 0, stream>>>(
        h2, W3, nullptr, nullptr, h3, b3, nullptr, nullptr,
        MROWS, 768, 512, 0, 768);

    // vocab projection: HBM-store-bound; nontemporal f32 stores
    gemm_bf16_wmma_kernel<0, 2><<<dim3((VOCAB + 63) / 64, MROWS / 128), 256, 0, stream>>>(
        h3, W4, nullptr, logits, nullptr, b4, nullptr, nullptr,
        MROWS, VOCAB, 768, 0, VOCAB);
}
